// RoPEMultiHeadAttention_82755429859947
// MI455X (gfx1250) — matrix operand order repaired and checked by reading, not compiled, not run on hardware
//
#include <hip/hip_runtime.h>
#include <hip/hip_bf16.h>
#include <cstdint>

// ---------------- common types / helpers ----------------
typedef __attribute__((ext_vector_type(16))) __bf16 v16bf;
typedef __attribute__((ext_vector_type(8)))  float  v8f;

union Frag16 { v16bf v; uint4 u[2]; };

static constexpr int E_DIM = 1024;
static constexpr int NHEAD = 16;
static constexpr int HDIM  = 64;
static constexpr int SEQ   = 2048;
static constexpr int BATCH = 4;
static constexpr int MTOT  = BATCH * SEQ;   // 8192 rows

__device__ __forceinline__ unsigned short bf16_bits(float f) {
  unsigned u = __builtin_bit_cast(unsigned, f);
  unsigned r = u + 0x7FFFu + ((u >> 16) & 1u);   // round-to-nearest-even
  return (unsigned short)(r >> 16);
}
__device__ __forceinline__ __bf16 f2bf(float f) {
  unsigned short h = bf16_bits(f);
  return __builtin_bit_cast(__bf16, h);
}
__device__ __forceinline__ unsigned pk2(float a, float b) {
  return (unsigned)bf16_bits(a) | ((unsigned)bf16_bits(b) << 16);
}

// ---------------- async global->LDS copy (CDNA5 ASYNCcnt path) -------------
#if __has_builtin(__builtin_amdgcn_global_load_async_to_lds_b128)
#define ASYNC_LDS 1
typedef int v4i __attribute__((vector_size(16)));
typedef __attribute__((address_space(1))) v4i gv4i;   // global
typedef __attribute__((address_space(3))) v4i lv4i;   // LDS
__device__ __forceinline__ void async_copy_b128(const void* g, void* l) {
  // direct pointer casts -> real addrspacecast (flat->LDS translation)
  __builtin_amdgcn_global_load_async_to_lds_b128((gv4i*)g, (lv4i*)l, 0, 0);
}
#else
#define ASYNC_LDS 0
__device__ __forceinline__ void async_copy_b128(const void* g, void* l) {
  *(uint4*)l = *(const uint4*)g;
}
#endif
__device__ __forceinline__ void wait_async0() {
#if ASYNC_LDS
#if __has_builtin(__builtin_amdgcn_s_wait_asynccnt)
  __builtin_amdgcn_s_wait_asynccnt(0);
#else
  asm volatile("s_wait_asynccnt 0" ::: "memory");
#endif
#endif
}

// ---------------- WMMA GEMM: Y[M,N] = X[M,K] * W[N,K]^T + b ----------------
// M=8192, N=1024, K=1024.  Block tile 128x128, 8 waves (4M x 2N),
// wave tile 32x64 (2 Mtiles x 4 Ntiles), K step 32, bf16 LDS staging.
// MODE 0: store bf16 [B,H,S,D]          (V projection)
// MODE 1: store bf16 [B,H,S,D] + RoPE   (Q,K projections)
// MODE 2: store fp32 [M,N]              (output projection)
template<bool IN_BF16, int MODE>
__global__ __launch_bounds__(256)
void gemm_wmma(const void* __restrict__ Xin, const float* __restrict__ W,
               const float* __restrict__ bias, void* __restrict__ Out)
{
  constexpr int LDT = 40;                 // padded K-stride (bf16 elems)
  __shared__ __bf16 xs[128 * LDT];
  __shared__ __bf16 wsh[128 * LDT];

  const int tid    = threadIdx.x;
  const int lane   = tid & 31;
  const int wv     = tid >> 5;
  const int wm     = wv >> 1;             // 0..3
  const int wn     = wv & 1;              // 0..1
  const int mBase  = blockIdx.y * 128;
  const int nBase  = blockIdx.x * 128;
  const int mWave  = wm * 32;             // block-local
  const int nWave  = wn * 64;             // head-aligned (64-wide)
  const int lane_n = lane & 15;
  const int halfk  = lane >> 4;

  v8f acc[2][4];
#pragma unroll
  for (int i = 0; i < 2; i++)
#pragma unroll
    for (int j = 0; j < 4; j++) acc[i][j] = (v8f)0.0f;

  const int lr = tid >> 1;                // 0..127 tile row
  const int lc = (tid & 1) * 16;          // 0 or 16 (K offset)

  for (int kk = 0; kk < E_DIM; kk += 32) {
    // ---- stage X tile (128x32) as bf16 ----
    bool used_async = false;
    if constexpr (IN_BF16) {
      const __bf16* src = (const __bf16*)Xin + (size_t)(mBase + lr) * E_DIM + kk + lc;
      __bf16* dst = xs + lr * LDT + lc;
      async_copy_b128(src, dst);              // bf16 already: DMA straight to LDS
      async_copy_b128(src + 8, dst + 8);
      used_async = true;
    } else {
      const float4* src = (const float4*)((const float*)Xin +
                           (size_t)(mBase + lr) * E_DIM + kk + lc);
      float4 a = src[0], b = src[1], c = src[2], d = src[3];
      uint4* dst = (uint4*)(xs + lr * LDT + lc);
      dst[0] = make_uint4(pk2(a.x,a.y), pk2(a.z,a.w), pk2(b.x,b.y), pk2(b.z,b.w));
      dst[1] = make_uint4(pk2(c.x,c.y), pk2(c.z,c.w), pk2(d.x,d.y), pk2(d.z,d.w));
    }
    // ---- stage W tile (128 output-cols x 32) as bf16 (fp32 -> convert) ----
    {
      const float4* src = (const float4*)(W + (size_t)(nBase + lr) * E_DIM + kk + lc);
      float4 a = src[0], b = src[1], c = src[2], d = src[3];
      uint4* dst = (uint4*)(wsh + lr * LDT + lc);
      dst[0] = make_uint4(pk2(a.x,a.y), pk2(a.z,a.w), pk2(b.x,b.y), pk2(b.z,b.w));
      dst[1] = make_uint4(pk2(c.x,c.y), pk2(c.z,c.w), pk2(d.x,d.y), pk2(d.z,d.w));
    }
    // prefetch next K-step rows into cache while this tile is consumed
    if (kk + 32 < E_DIM) {
      if constexpr (IN_BF16)
        __builtin_prefetch((const __bf16*)Xin + (size_t)(mBase + lr) * E_DIM + kk + 32 + lc, 0, 3);
      else
        __builtin_prefetch((const float*)Xin + (size_t)(mBase + lr) * E_DIM + kk + 32 + lc, 0, 3);
      __builtin_prefetch(W + (size_t)(nBase + lr) * E_DIM + kk + 32 + lc, 0, 3);
    }
    if (used_async) wait_async0();
    __syncthreads();

    // ---- fragments per documented CDNA5 16-bit layouts ----
    Frag16 af[2], bfv[4];
#pragma unroll
    for (int mt = 0; mt < 2; mt++) {
      const __bf16* p = xs + (mWave + mt * 16 + lane_n) * LDT;
      af[mt].u[0] = *(const uint4*)(p + halfk * 8);        // VGPR0-3: K=half*8..+7
      af[mt].u[1] = *(const uint4*)(p + 16 + halfk * 8);   // VGPR4-7: K=16+half*8..+7
    }
#pragma unroll
    for (int nt = 0; nt < 4; nt++) {
      const __bf16* p = wsh + (nWave + nt * 16 + lane_n) * LDT + halfk * 8;
      bfv[nt].u[0] = *(const uint4*)(p);
      bfv[nt].u[1] = *(const uint4*)(p + 16);
    }
#pragma unroll
    for (int mt = 0; mt < 2; mt++)
#pragma unroll
      for (int nt = 0; nt < 4; nt++)
        acc[mt][nt] = __builtin_amdgcn_wmma_f32_16x16x32_bf16(
            false, af[mt].v, false, bfv[nt].v, (short)0, acc[mt][nt], false, false);
    __syncthreads();
  }

  // ---- epilogue ----
  const int half = lane >> 4;
  float bn[4];
#pragma unroll
  for (int nt = 0; nt < 4; nt++) bn[nt] = bias[nBase + nWave + nt * 16 + lane_n];

  if constexpr (MODE == 2) {
    float* out = (float*)Out;
#pragma unroll
    for (int mt = 0; mt < 2; mt++)
#pragma unroll
      for (int r = 0; r < 8; r++) {
        int m = mBase + mWave + mt * 16 + half * 8 + r;
#pragma unroll
        for (int nt = 0; nt < 4; nt++) {
          int n = nBase + nWave + nt * 16 + lane_n;
          out[(size_t)m * E_DIM + n] = acc[mt][nt][r] + bn[nt];
        }
      }
  } else {
    __bf16* outw = (__bf16*)Out;
    const int h = (nBase + nWave) / HDIM;     // wave N range == one head
#pragma unroll
    for (int mt = 0; mt < 2; mt++) {
#pragma unroll
      for (int r = 0; r < 8; r++) {
        int m = mBase + mWave + mt * 16 + half * 8 + r;
        int b = m >> 11, s = m & 2047;
        float f[4];
#pragma unroll
        for (int nt = 0; nt < 4; nt++) f[nt] = acc[mt][nt][r] + bn[nt];
        if constexpr (MODE == 1) {
          // RoPE: partner column (d +/- 32) is the +/-2 N-tile, same lane.
          float g[4];
#pragma unroll
          for (int nt = 0; nt < 4; nt++) {
            int d  = nt * 16 + lane_n;
            int dj = d & 31;
            float theta = (float)s * __expf(-(float)dj * (9.2103403720f / 32.0f));
            float sn, cs;
            __sincosf(theta, &sn, &cs);
            g[nt] = (nt < 2) ? (f[nt] * cs - f[nt + 2] * sn)
                             : (f[nt] * cs + f[nt - 2] * sn);
          }
#pragma unroll
          for (int nt = 0; nt < 4; nt++) f[nt] = g[nt];
        }
#pragma unroll
        for (int nt = 0; nt < 4; nt++) {
          int d = nt * 16 + lane_n;
          outw[(((size_t)(b * NHEAD + h)) * SEQ + s) * HDIM + d] = f2bf(f[nt]);
        }
      }
    }
  }
}

// ---------------- causal flash attention (bf16 WMMA, fp32 softmax) ----------
// grid: (B*H = 64, S/128 = 16); block 256 threads = 8 waves, wave = 16 q rows.
__global__ __launch_bounds__(256)
void attn_wmma(const __bf16* __restrict__ Q, const __bf16* __restrict__ Kb,
               const __bf16* __restrict__ V, __bf16* __restrict__ O)
{
  constexpr int LDK = 72;                       // padded stride (bf16 elems)
  __shared__ __bf16 kt[64 * LDK];               // [key][d]
  __shared__ __bf16 vt[64 * LDK];               // transposed [d][key]
  __shared__ __bf16 pb[8][16 * LDK];            // per-wave P scratch [row][key]

  const int tid    = threadIdx.x;
  const int lane   = tid & 31;
  const int wv     = tid >> 5;
  const int bh     = blockIdx.x;                // 0..63
  const int qBase  = blockIdx.y * 128;
  const int q0     = qBase + wv * 16;
  const int lane_n = lane & 15;
  const int halfk  = lane >> 4;

  const size_t base = (size_t)bh * SEQ * HDIM;

  // Q fragments (kept in registers for whole key loop)
  Frag16 qf[2];
#pragma unroll
  for (int ks = 0; ks < 2; ks++) {
    const __bf16* p = Q + base + (size_t)(q0 + lane_n) * HDIM + ks * 32 + halfk * 8;
    qf[ks].u[0] = *(const uint4*)(p);
    qf[ks].u[1] = *(const uint4*)(p + 16);
  }

  v8f oacc[4];
#pragma unroll
  for (int nt = 0; nt < 4; nt++) oacc[nt] = (v8f)0.0f;
  float rowm[8], rowl[8];
#pragma unroll
  for (int r = 0; r < 8; r++) { rowm[r] = -1e30f; rowl[r] = 0.0f; }

  const int lr  = tid >> 2;            // 0..63 key row
  const int seg = (tid & 3) * 16;      // d offset

  const int nTiles = (qBase + 128) >> 6;
  for (int ktb = 0; ktb < nTiles; ktb++) {
    const int k0 = ktb * 64;
    // ---- K tile: async DMA to LDS (bf16, natural [key][d] layout) ----
    {
      const __bf16* ksrc = Kb + base + (size_t)(k0 + lr) * HDIM + seg;
      __bf16* kdst = kt + lr * LDK + seg;
      async_copy_b128(ksrc, kdst);
      async_copy_b128(ksrc + 8, kdst + 8);

      // ---- V tile: load + transpose into [d][key] (needs VGPR pass) ----
      union { uint4 u[2]; __bf16 e[16]; } vvv;
      const uint4* vsrc = (const uint4*)(V + base + (size_t)(k0 + lr) * HDIM + seg);
      vvv.u[0] = vsrc[0]; vvv.u[1] = vsrc[1];
#pragma unroll
      for (int j = 0; j < 16; j++) vt[(seg + j) * LDK + lr] = vvv.e[j];
    }
    wait_async0();
    __syncthreads();

    if (k0 <= q0 + 15) {               // tile has at least one unmasked key
      // ---- S = Q K^T (fp32 accum) ----
      v8f sacc[4];
#pragma unroll
      for (int nt = 0; nt < 4; nt++) sacc[nt] = (v8f)0.0f;
#pragma unroll
      for (int ks = 0; ks < 2; ks++) {
#pragma unroll
        for (int nt = 0; nt < 4; nt++) {
          Frag16 bfv;
          const __bf16* p = kt + (nt * 16 + lane_n) * LDK + ks * 32 + halfk * 8;
          bfv.u[0] = *(const uint4*)(p);
          bfv.u[1] = *(const uint4*)(p + 16);
          sacc[nt] = __builtin_amdgcn_wmma_f32_16x16x32_bf16(
              false, qf[ks].v, false, bfv.v, (short)0, sacc[nt], false, false);
        }
      }
      // ---- scale + causal mask ----
#pragma unroll
      for (int nt = 0; nt < 4; nt++) {
        int key = k0 + nt * 16 + lane_n;
#pragma unroll
        for (int r = 0; r < 8; r++) {
          int srow = q0 + halfk * 8 + r;
          float v = sacc[nt][r] * 0.125f;          // 1/sqrt(64)
          if (key > srow) v = -1e30f;
          sacc[nt][r] = v;
        }
      }
      // ---- online softmax (rows live across 16 lanes, xor<16 stays in half) ----
#pragma unroll
      for (int r = 0; r < 8; r++) {
        float mx = fmaxf(fmaxf(sacc[0][r], sacc[1][r]),
                         fmaxf(sacc[2][r], sacc[3][r]));
#pragma unroll
        for (int off = 1; off < 16; off <<= 1) mx = fmaxf(mx, __shfl_xor(mx, off, 32));
        float nm = fmaxf(rowm[r], mx);
        float sc = __expf(rowm[r] - nm);
        float sum = 0.0f;
#pragma unroll
        for (int nt = 0; nt < 4; nt++) {
          float p = __expf(sacc[nt][r] - nm);
          sacc[nt][r] = p;
          sum += p;
        }
#pragma unroll
        for (int off = 1; off < 16; off <<= 1) sum += __shfl_xor(sum, off, 32);
        rowl[r] = rowl[r] * sc + sum;
        rowm[r] = nm;
#pragma unroll
        for (int nt = 0; nt < 4; nt++) oacc[nt][r] *= sc;
      }
      // ---- P -> LDS (re-layout C/D -> A fragment), then O += P V ----
#pragma unroll
      for (int nt = 0; nt < 4; nt++)
#pragma unroll
        for (int r = 0; r < 8; r++)
          pb[wv][(halfk * 8 + r) * LDK + nt * 16 + lane_n] = f2bf(sacc[nt][r]);

#pragma unroll
      for (int ks = 0; ks < 2; ks++) {
        Frag16 pf;
        const __bf16* pp = &pb[wv][lane_n * LDK + ks * 32 + halfk * 8];
        pf.u[0] = *(const uint4*)(pp);
        pf.u[1] = *(const uint4*)(pp + 16);
#pragma unroll
        for (int nt = 0; nt < 4; nt++) {
          Frag16 vf;
          const __bf16* vp = vt + (nt * 16 + lane_n) * LDK + ks * 32 + halfk * 8;
          vf.u[0] = *(const uint4*)(vp);
          vf.u[1] = *(const uint4*)(vp + 16);
          oacc[nt] = __builtin_amdgcn_wmma_f32_16x16x32_bf16(
              false, pf.v, false, vf.v, (short)0, oacc[nt], false, false);
        }
      }
    }
    __syncthreads();
  }

  // ---- normalize + store attn out as bf16 [B,S,E] ----
  const int b = bh >> 4, h = bh & 15;
#pragma unroll
  for (int r = 0; r < 8; r++) {
    int srow = q0 + halfk * 8 + r;
    float inv = 1.0f / rowl[r];
#pragma unroll
    for (int nt = 0; nt < 4; nt++) {
      int d = nt * 16 + lane_n;
      O[((size_t)(b * SEQ + srow)) * E_DIM + h * HDIM + d] = f2bf(oacc[nt][r] * inv);
    }
  }
}

// ---------------- host launch ----------------
extern "C" void kernel_launch(void* const* d_in, const int* in_sizes, int n_in,
                              void* d_out, int out_size, void* d_ws, size_t ws_size,
                              hipStream_t stream) {
  const float* q  = (const float*)d_in[0];
  const float* k  = (const float*)d_in[1];
  const float* v  = (const float*)d_in[2];
  const float* Wq = (const float*)d_in[3];
  const float* bq = (const float*)d_in[4];
  const float* Wk = (const float*)d_in[5];
  const float* bk = (const float*)d_in[6];
  const float* Wv = (const float*)d_in[7];
  const float* bv = (const float*)d_in[8];
  const float* Wo = (const float*)d_in[9];
  const float* bo = (const float*)d_in[10];

  const size_t elems = (size_t)BATCH * NHEAD * SEQ * HDIM;   // 8,388,608
  __bf16* qws = (__bf16*)d_ws;
  __bf16* kws = qws + elems;
  __bf16* vws = kws + elems;
  __bf16* aws = vws + elems;

  dim3 gblk(E_DIM / 128, MTOT / 128);   // (8, 64)
  dim3 thr(256);

  gemm_wmma<false, 1><<<gblk, thr, 0, stream>>>(q, Wq, bq, qws);   // Q + RoPE
  gemm_wmma<false, 1><<<gblk, thr, 0, stream>>>(k, Wk, bk, kws);   // K + RoPE
  gemm_wmma<false, 0><<<gblk, thr, 0, stream>>>(v, Wv, bv, vws);   // V

  attn_wmma<<<dim3(BATCH * NHEAD, SEQ / 128), thr, 0, stream>>>(qws, kws, vws, aws);

  gemm_wmma<true, 2><<<gblk, thr, 0, stream>>>(aws, Wo, bo, (float*)d_out);
}
